// MultiHeadAttention_40939628265522
// MI455X (gfx1250) — matrix operand order repaired and checked by reading, not compiled, not run on hardware
//
#include <hip/hip_runtime.h>

typedef __attribute__((ext_vector_type(16))) __bf16 v16bf;
typedef __attribute__((ext_vector_type(8)))  float  v8f;

#define TS 40  // padded LDS row stride in ushorts (80B, 16B-aligned, bank-friendly)

union BFV {
  v16bf v;
  uint4 q[2];
  unsigned short u[16];
};

__device__ __forceinline__ unsigned short f2bf(float f) {
  union { float f; unsigned int u; } x; x.f = f;
  unsigned int r = x.u + 0x7FFFu + ((x.u >> 16) & 1u);  // round-to-nearest-even
  return (unsigned short)(r >> 16);
}

__device__ __forceinline__ unsigned pk2bf(float lo, float hi) {
  return (unsigned)f2bf(lo) | ((unsigned)f2bf(hi) << 16);
}

// Async 16B copy global -> LDS (CDNA5 GLOBAL_LOAD_ASYNC_TO_LDS_B128, ASYNCcnt).
// lds_off = LDS byte offset (low 32 bits of the flat shared pointer).
__device__ __forceinline__ void async_copy_b128(unsigned lds_off, const void* gptr) {
  unsigned long long ga = (unsigned long long)gptr;
  asm volatile("global_load_async_to_lds_b128 %0, %1, off"
               :: "v"(lds_off), "v"(ga) : "memory");
}

__device__ __forceinline__ void wait_async0() {
  asm volatile("s_wait_asynccnt 0x0" ::: "memory");
}

// ---------------------------------------------------------------------------
// Kernel 1: padding masks. kmask/qmask[b*S+s] = sign(sum_d |x[b,s,d]|)
// ---------------------------------------------------------------------------
__global__ void masks_kernel(const float* __restrict__ key,
                             const float* __restrict__ query,
                             float* __restrict__ kmask,
                             float* __restrict__ qmask) {
  const int lane = threadIdx.x & 31;
  const int wid  = threadIdx.x >> 5;
  const int row  = blockIdx.x * 8 + wid;  // 0..8191
  const float* kp = key   + (size_t)row * 256;
  const float* qp = query + (size_t)row * 256;
  float sk = 0.f, sq = 0.f;
#pragma unroll
  for (int i = 0; i < 8; ++i) {
    sk += fabsf(kp[lane + i * 32]);
    sq += fabsf(qp[lane + i * 32]);
  }
  for (int off = 16; off > 0; off >>= 1) {
    sk += __shfl_xor(sk, off, 32);
    sq += __shfl_xor(sq, off, 32);
  }
  if (lane == 0) {
    kmask[row] = (sk > 0.f) ? 1.f : 0.f;
    qmask[row] = (sq > 0.f) ? 1.f : 0.f;
  }
}

// ---------------------------------------------------------------------------
// Kernel 2: projection GEMM  out[n,j] = sum_k X[n,k] * W[j,k] + b[j]
// blockIdx.z selects Q/K/V. Output bf16 head-major [B,H,S,DH]; V transposed
// to [B,H,DH,S]. WG = 256 threads (8 waves), tile 64(M) x 128(N), K-step 32.
// Staging: b128 global loads, packed bf16 convert, b128 DS stores.
// ---------------------------------------------------------------------------
__global__ void proj_kernel(const float* __restrict__ q_in,
                            const float* __restrict__ k_in,
                            const float* __restrict__ v_in,
                            const float* __restrict__ Wq, const float* __restrict__ bq,
                            const float* __restrict__ Wk, const float* __restrict__ bk,
                            const float* __restrict__ Wv, const float* __restrict__ bv,
                            unsigned short* __restrict__ qh,
                            unsigned short* __restrict__ kh,
                            unsigned short* __restrict__ vt) {
  const int S = 2048, D = 256, H = 8, DH = 32;
  __shared__ alignas(16) unsigned short xt[64 * TS];
  __shared__ alignas(16) unsigned short wt[128 * TS];

  const float* X; const float* W; const float* bias; unsigned short* dst; int transpose;
  if (blockIdx.z == 0)      { X = q_in; W = Wq; bias = bq; dst = qh; transpose = 0; }
  else if (blockIdx.z == 1) { X = k_in; W = Wk; bias = bk; dst = kh; transpose = 0; }
  else                      { X = v_in; W = Wv; bias = bv; dst = vt; transpose = 1; }

  const int row0 = blockIdx.x * 64;   // M origin (rows of X)
  const int col0 = blockIdx.y * 128;  // N origin (output feature j)
  const int tid  = threadIdx.x;
  const int lane = tid & 31, wid = tid >> 5;
  const int half = lane >> 4, l16 = lane & 15;
  const int m0w = (wid & 3) * 16;   // wave's M sub-tile
  const int n0w = (wid >> 2) * 64;  // wave's N range (4 tiles of 16)

  v8f zero = {};
  v8f acc[4];
#pragma unroll
  for (int t = 0; t < 4; ++t) acc[t] = zero;

  for (int k0 = 0; k0 < D; k0 += 32) {
    __syncthreads();
    {  // stage X tile: 64 rows x 32 cols, 8 floats per thread, one b128 DS store
      int r = tid >> 2, c = (tid & 3) * 8;
      const float* src = &X[(size_t)(row0 + r) * D + k0 + c];
      float4 a0 = *(const float4*)(src);
      float4 a1 = *(const float4*)(src + 4);
      uint4 u;
      u.x = pk2bf(a0.x, a0.y); u.y = pk2bf(a0.z, a0.w);
      u.z = pk2bf(a1.x, a1.y); u.w = pk2bf(a1.z, a1.w);
      *(uint4*)&xt[r * TS + c] = u;
    }
#pragma unroll
    for (int it = 0; it < 2; ++it) {  // stage W tile: 128 rows x 32 cols
      int t = tid + it * 256;
      int r = t >> 2, c = (t & 3) * 8;
      const float* src = &W[(size_t)(col0 + r) * D + k0 + c];
      float4 a0 = *(const float4*)(src);
      float4 a1 = *(const float4*)(src + 4);
      uint4 u;
      u.x = pk2bf(a0.x, a0.y); u.y = pk2bf(a0.z, a0.w);
      u.z = pk2bf(a1.x, a1.y); u.w = pk2bf(a1.z, a1.w);
      *(uint4*)&wt[r * TS + c] = u;
    }
    __syncthreads();

    // A operand: row m = m0w+l16, K runs {kA..kA+8, kA+16..kA+24}, kA = half?8:0
    BFV a;
    {
      const unsigned short* p = &xt[(m0w + l16) * TS + (half ? 8 : 0)];
      a.q[0] = *(const uint4*)(p);
      a.q[1] = *(const uint4*)(p + 16);
    }
#pragma unroll
    for (int t = 0; t < 4; ++t) {
      // B operand: col j = n0w+t*16+l16, K runs {kB..kB+8, kB+16..kB+24}, kB = half?8:0
      BFV bmat;
      const unsigned short* p = &wt[(n0w + t * 16 + l16) * TS + (half ? 8 : 0)];
      bmat.q[0] = *(const uint4*)(p);
      bmat.q[1] = *(const uint4*)(p + 16);
      acc[t] = __builtin_amdgcn_wmma_f32_16x16x32_bf16(
          false, a.v, false, bmat.v, (short)0, acc[t], false, false);
    }
  }

  // Epilogue: bias, bf16 convert, scatter to head-major (or transposed for V)
#pragma unroll
  for (int t = 0; t < 4; ++t) {
    int j  = col0 + n0w + t * 16 + l16;
    float bj = bias[j];
    int h = j >> 5, dh = j & 31;
#pragma unroll
    for (int v = 0; v < 8; ++v) {
      int n = row0 + m0w + v + 8 * half;  // C layout: lanes 0-15 -> M=v, 16-31 -> M=v+8
      int b = n >> 11, s = n & 2047;
      unsigned short val = f2bf(acc[t][v] + bj);
      if (!transpose)
        dst[((size_t)(b * H + h) * S + s) * DH + dh] = val;
      else
        dst[((size_t)(b * H + h) * DH + dh) * S + s] = val;
    }
  }
}

// ---------------------------------------------------------------------------
// Kernel 3: causal flash attention. WG = 256 threads = 8 waves per (b,h,
// 128-q-row tile). Each wave owns 16 q rows; Q in registers as A operand.
// K/V^T tiles staged with GLOBAL_LOAD_ASYNC_TO_LDS_B128 (ASYNCcnt), next
// tiles prefetched. Per 32-k-step: 2 WMMAs Q*K^T, online softmax with
// shuffle reductions, P via LDS, 2 WMMAs P*V.
// ---------------------------------------------------------------------------
__global__ void attn_kernel(const unsigned short* __restrict__ qh,
                            const unsigned short* __restrict__ kh,
                            const unsigned short* __restrict__ vt,
                            const float* __restrict__ kmask,
                            const float* __restrict__ qmask,
                            const float* __restrict__ query,
                            float* __restrict__ out) {
  const int S = 2048, D = 256, H = 8, DH = 32;
  const float SCALE = 0.17677669529663688f;  // 1/sqrt(32)
  const float NEG   = -1e9f;

  __shared__ alignas(16) unsigned short kbuf[32 * TS];       // K tile [kpos][d]
  __shared__ alignas(16) unsigned short vbuf[32 * TS];       // V tile [d][kpos]
  __shared__ alignas(16) unsigned short pbuf[8 * 16 * TS];   // per-wave P tiles

  const int q0 = blockIdx.x * 128;
  const int h  = blockIdx.y;
  const int b  = blockIdx.z;
  const size_t bh = (size_t)(b * H + h);

  const int tid  = threadIdx.x, lane = tid & 31, wid = tid >> 5;
  const int half = lane >> 4, l16 = lane & 15;

  // Per-thread async-copy source/dest for staging (wave-uniform branch: waves
  // 0-3 copy the K tile, waves 4-7 the V^T tile; 16B per lane).
  const int ct  = tid & 127;       // 0..127
  const int cr  = ct >> 2;         // tile row 0..31
  const int cq  = (ct & 3) * 8;    // ushort col 0,8,16,24 (16B quarters)
  const bool do_k = (wid < 4);
  unsigned lds_off;
  const unsigned short* gsrc0;     // source for kbase = 0
  size_t gstep;                    // source advance per 32-k step (in ushorts)
  if (do_k) {
    lds_off = (unsigned)(size_t)(const void*)&kbuf[cr * TS + cq];
    gsrc0   = kh + (bh * S + cr) * DH + cq;
    gstep   = (size_t)32 * DH;     // 32 seq rows forward (in ushorts)
  } else {
    lds_off = (unsigned)(size_t)(const void*)&vbuf[cr * TS + cq];
    gsrc0   = vt + (bh * DH + cr) * S + cq;
    gstep   = 32;                  // 32 columns forward (in ushorts)
  }

  // Q A-operand for this wave's 16 rows (contraction dim = DH = 32)
  BFV qa;
  {
    const unsigned short* qp =
        qh + (bh * S + q0 + wid * 16 + l16) * DH + (half ? 8 : 0);
    qa.q[0] = *(const uint4*)(qp);
    qa.q[1] = *(const uint4*)(qp + 16);
  }

  v8f zero = {};
  v8f accA = zero, accB = zero;  // output cols d=0..15 / d=16..31
  float mrow[8], lrow[8];
#pragma unroll
  for (int v = 0; v < 8; ++v) { mrow[v] = -3e38f; lrow[v] = 0.f; }

  const int kend = q0 + 128;  // causal: no k tiles beyond the q tile
  for (int kbase = 0; kbase < kend; kbase += 32) {
    __syncthreads();  // previous iteration done reading kbuf/vbuf
    {
      const unsigned short* src = gsrc0 + (size_t)(kbase >> 5) * gstep;
      async_copy_b128(lds_off, src);
      if (kbase + 32 < kend)  // prefetch next tile into L2/L0
        __builtin_prefetch(src + gstep, 0, 0);
    }
    wait_async0();
    __syncthreads();

    // Q*K^T: two 16x16 score tiles (k cols 0..15 and 16..31 of this step)
    v8f s0 = zero, s1 = zero;
    {
      BFV kb;
      const unsigned short* p = &kbuf[l16 * TS + (half ? 8 : 0)];
      kb.q[0] = *(const uint4*)p; kb.q[1] = *(const uint4*)(p + 16);
      s0 = __builtin_amdgcn_wmma_f32_16x16x32_bf16(false, qa.v, false, kb.v,
                                                   (short)0, s0, false, false);
      const unsigned short* p2 = &kbuf[(16 + l16) * TS + (half ? 8 : 0)];
      kb.q[0] = *(const uint4*)p2; kb.q[1] = *(const uint4*)(p2 + 16);
      s1 = __builtin_amdgcn_wmma_f32_16x16x32_bf16(false, qa.v, false, kb.v,
                                                   (short)0, s1, false, false);
    }

    const int kc0 = kbase + l16;
    const int kc1 = kbase + 16 + l16;
    const float km0 = kmask[b * S + kc0];
    const float km1 = kmask[b * S + kc1];

    // Online softmax across the 16x32 block; rows live as (vgpr, lane-half).
#pragma unroll
    for (int v = 0; v < 8; ++v) {
      int qrow = q0 + wid * 16 + v + 8 * half;
      float x0 = (km0 != 0.f && kc0 <= qrow) ? s0[v] * SCALE : NEG;
      float x1 = (km1 != 0.f && kc1 <= qrow) ? s1[v] * SCALE : NEG;
      float sm = fmaxf(x0, x1);  // row-max over the 16 lanes of this half
      sm = fmaxf(sm, __shfl_xor(sm, 1, 32));
      sm = fmaxf(sm, __shfl_xor(sm, 2, 32));
      sm = fmaxf(sm, __shfl_xor(sm, 4, 32));
      sm = fmaxf(sm, __shfl_xor(sm, 8, 32));
      float mnew  = fmaxf(mrow[v], sm);
      float alpha = __expf(mrow[v] - mnew);
      float p0 = __expf(x0 - mnew);
      float p1 = __expf(x1 - mnew);
      float rs = p0 + p1;
      rs += __shfl_xor(rs, 1, 32);
      rs += __shfl_xor(rs, 2, 32);
      rs += __shfl_xor(rs, 4, 32);
      rs += __shfl_xor(rs, 8, 32);
      lrow[v] = lrow[v] * alpha + rs;
      mrow[v] = mnew;
      accA[v] *= alpha;
      accB[v] *= alpha;
      int m = v + 8 * half;  // C-layout row -> P row
      pbuf[(wid * 16 + m) * TS + l16]      = f2bf(p0);
      pbuf[(wid * 16 + m) * TS + 16 + l16] = f2bf(p1);
    }
    __syncthreads();  // uniform barrier; makes P visible for A-layout reload

    // P*V: reload P in A-operand layout, V^T tile as B operand
    BFV pa;
    {
      const unsigned short* p = &pbuf[(wid * 16 + l16) * TS + (half ? 8 : 0)];
      pa.q[0] = *(const uint4*)p;
      pa.q[1] = *(const uint4*)(p + 16);
    }
    {
      BFV vb;
      const unsigned short* p = &vbuf[l16 * TS + (half ? 8 : 0)];
      vb.q[0] = *(const uint4*)p; vb.q[1] = *(const uint4*)(p + 16);
      accA = __builtin_amdgcn_wmma_f32_16x16x32_bf16(false, pa.v, false, vb.v,
                                                     (short)0, accA, false, false);
      const unsigned short* p2 = &vbuf[(16 + l16) * TS + (half ? 8 : 0)];
      vb.q[0] = *(const uint4*)p2; vb.q[1] = *(const uint4*)(p2 + 16);
      accB = __builtin_amdgcn_wmma_f32_16x16x32_bf16(false, pa.v, false, vb.v,
                                                     (short)0, accB, false, false);
    }
  }

  // Epilogue: softmax normalize, query-mask, residual, fp32 store
#pragma unroll
  for (int v = 0; v < 8; ++v) {
    int qrow = q0 + wid * 16 + v + 8 * half;
    float qm  = qmask[b * S + qrow];
    float inv = (lrow[v] > 0.f) ? qm / lrow[v] : 0.f;
    size_t o = (size_t)(b * S + qrow) * D + h * DH + l16;
    out[o]      = accA[v] * inv + query[o];
    out[o + 16] = accB[v] * inv + query[o + 16];
  }
}

// ---------------------------------------------------------------------------
extern "C" void kernel_launch(void* const* d_in, const int* in_sizes, int n_in,
                              void* d_out, int out_size, void* d_ws, size_t ws_size,
                              hipStream_t stream) {
  const float* query = (const float*)d_in[0];
  const float* key   = (const float*)d_in[1];
  const float* value = (const float*)d_in[2];
  const float* Wq    = (const float*)d_in[3];
  const float* bq    = (const float*)d_in[4];
  const float* Wk    = (const float*)d_in[5];
  const float* bk    = (const float*)d_in[6];
  const float* Wv    = (const float*)d_in[7];
  const float* bv    = (const float*)d_in[8];
  float* out = (float*)d_out;

  // Workspace: Qh, Kh [B,H,S,DH] bf16; Vt [B,H,DH,S] bf16; masks fp32.
  const size_t elems = (size_t)4 * 8 * 2048 * 32;  // 2,097,152 per tensor
  unsigned short* qh = (unsigned short*)d_ws;
  unsigned short* kh = qh + elems;
  unsigned short* vt = kh + elems;
  float* kmask = (float*)(vt + elems);
  float* qmask = kmask + 4 * 2048;

  masks_kernel<<<dim3(1024), 256, 0, stream>>>(key, query, kmask, qmask);
  proj_kernel<<<dim3(128, 2, 3), 256, 0, stream>>>(query, key, value,
                                                   Wq, bq, Wk, bk, Wv, bv,
                                                   qh, kh, vt);
  attn_kernel<<<dim3(16, 8, 4), 256, 0, stream>>>(qh, kh, vt, kmask, qmask,
                                                  query, out);
}
